// MatrixLoss_17858474016937
// MI455X (gfx1250) — matrix operand order repaired and checked by reading, not compiled, not run on hardware
//
#include <hip/hip_runtime.h>
#include <hip/hip_bf16.h>
#include <math.h>

#define S_    256
#define H_    1024
#define NTAG  20
#define NPAIR 32640   // 256*255/2

typedef __bf16 bf16;
typedef __attribute__((ext_vector_type(16))) __bf16 v16bf;
typedef __attribute__((ext_vector_type(8)))  __bf16 v8bf;
typedef __attribute__((ext_vector_type(8)))  float  v8f;

// ---------------- workspace layout (bytes) ----------------
#define WS_HID_BF16 0u         // 256*1024 bf16      = 524288
#define WS_W1T      524288u    // 1024*1024 bf16     = 2097152
#define WS_WTOPT    2621440u   // 64*1024 bf16       = 131072
#define WS_T        2752512u   // 256*1024 f32       = 1048576
#define WS_G        3801088u   // 256*64 f32         = 65536
#define WS_E        3866624u   // 256 f32
#define WS_ZC       3867648u   // 257 f32
#define WS_QC       3868928u   // 257*20 f32
#define WS_LTAB     3889664u   // 257*20 f32
#define WS_RUNID    3910400u   // 256 i32
#define WS_START    3911424u   // 256 i32
#define WS_END      3912448u   // 256 i32
#define WS_ACC      3913472u   // 1 f32

// ---------------- prep: f32 -> bf16 ----------------
__global__ void k_f32_to_bf16(const float* __restrict__ src, bf16* __restrict__ dst, int n){
  int i = blockIdx.x*blockDim.x + threadIdx.x;
  if (i < n) dst[i] = (bf16)src[i];
}

// tiled transpose + convert: dst[c*rows + r] = (bf16)src[r*cols + c]
__global__ void k_transpose_bf16(const float* __restrict__ src, bf16* __restrict__ dst,
                                 int rows, int cols){
  __shared__ float tile[32][33];
  int r0 = blockIdx.y*32, c0 = blockIdx.x*32;
  for (int rr = threadIdx.y; rr < 32; rr += blockDim.y){
    int r = r0 + rr, c = c0 + threadIdx.x;
    tile[rr][threadIdx.x] = (r < rows && c < cols) ? src[(long)r*cols + c] : 0.f;
  }
  __syncthreads();
  for (int cc = threadIdx.y; cc < 32; cc += blockDim.y){
    int c = c0 + cc, r = r0 + threadIdx.x;
    if (c < cols && r < rows) dst[(long)c*rows + r] = (bf16)tile[threadIdx.x][cc];
  }
}

// wtopT[n][k] = fc_w[(n/20)*1024 + k][n%20], n<60 ; zero-pad n=60..63. K-major for WMMA B.
__global__ void k_build_wtopT(const float* __restrict__ fc_w, bf16* __restrict__ wtopT){
  int idx = blockIdx.x*blockDim.x + threadIdx.x;
  if (idx >= 64*H_) return;
  int n = idx / H_, k = idx % H_;
  float v = 0.f;
  if (n < 3*NTAG){
    int blk = n / NTAG, t = n % NTAG;
    v = fc_w[(long)(blk*H_ + k)*NTAG + t];
  }
  wtopT[idx] = (bf16)v;
}

// ---------------- WMMA GEMM: C[M,N] = A[M,K] @ B[K,N] (B supplied K-major as BT[N][K]) ----
// one 16x16 tile per wave, 4 waves per block. Optional epilogue: tanh(x + bias[n]).
__global__ void k_wmma_gemm(const bf16* __restrict__ A, const bf16* __restrict__ BT,
                            float* __restrict__ C, int M, int N, int K,
                            const float* __restrict__ bias, int do_tanh){
  const int lane = threadIdx.x & 31;
  const int wave = threadIdx.x >> 5;
  const int n0 = (blockIdx.x*4 + wave)*16;
  const int m0 = blockIdx.y*16;
  if (n0 >= N || m0 >= M) return;           // wave-uniform: EXEC stays all-1s
  const int half = lane >> 4;
  const int l16  = lane & 15;
  const bf16* arow = A  + (long)(m0 + l16)*K;          // A frag: M=lane%16
  const bf16* brow = BT + (long)(n0 + l16)*K + 16*half; // B frag: N=lane%16, K=16*half+e
  v8f acc = {};
  for (int k0 = 0; k0 < K; k0 += 32){
    // A frag (matches B): lane half h: element e -> K=16h+e
    v16bf a = *(const v16bf*)(arow + k0 + 16*half);
    v16bf b = *(const v16bf*)(brow + k0);
    acc = __builtin_amdgcn_wmma_f32_16x16x32_bf16(false, a, false, b,
                                                  (short)0, acc, false, false);
  }
  const int mbase = m0 + 8*half;            // C/D: VGPR r -> M = r + 8*half, N = lane%16
  const int n = n0 + l16;
  #pragma unroll
  for (int r = 0; r < 8; ++r){
    float v = acc[r];
    if (do_tanh) v = tanhf(v + bias[n]);
    C[(long)(mbase + r)*N + n] = v;
  }
}

// ---------------- attention score + softmax prefix (one block of 256) ----------------
__global__ void k_score_softmax(const float* __restrict__ T, const float* __restrict__ w2,
                                const float* __restrict__ b2,
                                float* __restrict__ e, float* __restrict__ Zc){
  const int s = threadIdx.x;
  __shared__ float sm[S_];
  float acc = 0.f;
  const float* row = T + (long)s*H_;
  for (int k = 0; k < H_; k += 4){
    float4 v = *(const float4*)(row + k);
    float4 w = *(const float4*)(w2 + k);
    acc += v.x*w.x + v.y*w.y + v.z*w.z + v.w*w.w;
  }
  float a = acc + b2[0];
  sm[s] = a; __syncthreads();
  for (int off = 128; off > 0; off >>= 1){
    if (s < off) sm[s] = fmaxf(sm[s], sm[s+off]);
    __syncthreads();
  }
  float amax = sm[0]; __syncthreads();
  float ev = expf(a - amax);
  e[s] = ev;
  sm[s] = ev; __syncthreads();
  for (int off = 1; off < S_; off <<= 1){      // inclusive Hillis-Steele scan
    float v = (s >= off) ? sm[s-off] : 0.f;
    __syncthreads();
    sm[s] += v;
    __syncthreads();
  }
  Zc[s+1] = sm[s];
  if (s == 0) Zc[0] = 0.f;
}

// ---------------- Qc[j][t] = sum_{s<j} e[s]*G[s][t]  (G col t = hidden@W_span col t) ----
__global__ void k_qc(const float* __restrict__ e, const float* __restrict__ G,
                     float* __restrict__ Qc){
  int t = threadIdx.x;
  if (t >= NTAG) return;
  float acc = 0.f;
  Qc[t] = 0.f;
  for (int s = 0; s < S_; ++s){
    acc += e[s] * G[s*64 + t];
    Qc[(s+1)*NTAG + t] = acc;
  }
}

// ---------------- length-embedding table: Ltab[len][t] = len_emb[len]@W_len + fc_b -----
__global__ void k_ltab(const float* __restrict__ len_emb, const float* __restrict__ fc_w,
                       const float* __restrict__ fc_b, float* __restrict__ Ltab){
  int idx = blockIdx.x*blockDim.x + threadIdx.x;
  if (idx >= 257*NTAG) return;
  int len = idx / NTAG, t = idx % NTAG;
  float v = fc_b[t];
  #pragma unroll
  for (int d = 0; d < 10; ++d)
    v += len_emb[len*10 + d] * fc_w[(long)(3*H_ + d)*NTAG + t];
  Ltab[idx] = v;
}

// ---------------- run ids / start / end flags from target (one block of 256) ----------
__global__ void k_labels(const int* __restrict__ target, int* __restrict__ run_id,
                         int* __restrict__ startf, int* __restrict__ endf,
                         float* __restrict__ acc){
  const int s = threadIdx.x;
  __shared__ int sm[S_];
  int ts = target[s];
  int chg = (s == 0) || (ts != target[s-1]);
  startf[s] = chg;
  endf[s]   = (s == S_-1) || (ts != target[s+1]);
  sm[s] = chg; __syncthreads();
  for (int off = 1; off < S_; off <<= 1){
    int v = (s >= off) ? sm[s-off] : 0;
    __syncthreads();
    sm[s] += v;
    __syncthreads();
  }
  run_id[s] = sm[s];
  if (s == 0) *acc = 0.f;   // zero loss accumulator here (runs before pair kernel)
}

// ---------------- per-pair logits + log-softmax NLL accumulation ----------------
__global__ void k_pairs(const float* __restrict__ Zc, const float* __restrict__ Qc,
                        const float* __restrict__ G, const float* __restrict__ Ltab,
                        const int* __restrict__ run_id, const int* __restrict__ startf,
                        const int* __restrict__ endf, const int* __restrict__ target,
                        float* __restrict__ out, float* __restrict__ lossAcc){
  const int i = blockIdx.x;       // 0..255
  const int j = threadIdx.x;      // 0..255
  __shared__ float red[S_];
  float lp = 0.f;
  if (j > i){
    float Zs  = Zc[j+1] - Zc[i];
    float inv = 1.f / Zs;
    int len = j - i + 1;
    const float* Qj = Qc + (j+1)*NTAG;
    const float* Qi = Qc + i*NTAG;
    const float* Ai = G + i*64 + NTAG;       // cols 20..39 = hidden@W_i
    const float* Bj = G + j*64 + 2*NTAG;     // cols 40..59 = hidden@W_j
    const float* Lr = Ltab + len*NTAG;
    int p = i*(2*S_ - i - 1)/2 + (j - i - 1);
    float* op = out + (long)p*NTAG;
    float lg[NTAG];
    float mx = -1e30f;
    #pragma unroll
    for (int t = 0; t < NTAG; ++t){
      float v = (Qj[t] - Qi[t])*inv + Ai[t] + Bj[t] + Lr[t];
      lg[t] = v;
      op[t] = v;
      mx = fmaxf(mx, v);
    }
    int label = (run_id[i] == run_id[j] && startf[i] && endf[j]) ? target[j] : 0;
    float se = 0.f, picked = 0.f;
    #pragma unroll
    for (int t = 0; t < NTAG; ++t){
      se += expf(lg[t] - mx);
      if (t == label) picked = lg[t];
    }
    lp = picked - (logf(se) + mx);
  }
  red[j] = lp; __syncthreads();
  for (int off = 128; off > 0; off >>= 1){
    if (j < off) red[j] += red[j+off];
    __syncthreads();
  }
  if (j == 0) atomicAdd(lossAcc, red[0]);
}

__global__ void k_finalize(const float* __restrict__ acc, float* __restrict__ out){
  out[(long)NPAIR*NTAG] = -(*acc) / (float)NPAIR;
}

// ---------------- host launch ----------------
extern "C" void kernel_launch(void* const* d_in, const int* in_sizes, int n_in,
                              void* d_out, int out_size, void* d_ws, size_t ws_size,
                              hipStream_t stream){
  const float* hidden  = (const float*)d_in[0];
  const int*   target  = (const int*)  d_in[1];
  const float* attn_w1 = (const float*)d_in[2];
  const float* attn_b1 = (const float*)d_in[3];
  const float* attn_w2 = (const float*)d_in[4];
  const float* attn_b2 = (const float*)d_in[5];
  const float* len_emb = (const float*)d_in[6];
  const float* fc_w    = (const float*)d_in[7];
  const float* fc_b    = (const float*)d_in[8];
  float* out = (float*)d_out;
  char*  ws  = (char*)d_ws;

  bf16*  hid_bf = (bf16*)(ws + WS_HID_BF16);
  bf16*  w1T    = (bf16*)(ws + WS_W1T);
  bf16*  wtopT  = (bf16*)(ws + WS_WTOPT);
  float* T      = (float*)(ws + WS_T);
  float* G      = (float*)(ws + WS_G);
  float* e      = (float*)(ws + WS_E);
  float* Zc     = (float*)(ws + WS_ZC);
  float* Qc     = (float*)(ws + WS_QC);
  float* Ltab   = (float*)(ws + WS_LTAB);
  int*   runid  = (int*)(ws + WS_RUNID);
  int*   startf = (int*)(ws + WS_START);
  int*   endf   = (int*)(ws + WS_END);
  float* acc    = (float*)(ws + WS_ACC);

  // prep
  k_f32_to_bf16<<<(S_*H_ + 255)/256, 256, 0, stream>>>(hidden, hid_bf, S_*H_);
  k_transpose_bf16<<<dim3(H_/32, H_/32), dim3(32, 8), 0, stream>>>(attn_w1, w1T, H_, H_);
  k_build_wtopT<<<(64*H_ + 255)/256, 256, 0, stream>>>(fc_w, wtopT);

  // GEMM1: T = tanh(hidden @ W1 + b1)   [256 x 1024]
  k_wmma_gemm<<<dim3(H_/64, S_/16), 128, 0, stream>>>(hid_bf, w1T, T, S_, H_, H_, attn_b1, 1);

  // scores a, e = exp(a - max), Zc prefix sums
  k_score_softmax<<<1, S_, 0, stream>>>(T, attn_w2, attn_b2, e, Zc);

  // GEMM2: G = hidden @ [W_span | W_i | W_j] (padded to 64 cols)   [256 x 64]
  k_wmma_gemm<<<dim3(1, S_/16), 128, 0, stream>>>(hid_bf, wtopT, G, S_, 64, H_, nullptr, 0);

  // Qc prefix sums, length table, labels (+ zero loss accumulator)
  k_qc<<<1, 32, 0, stream>>>(e, G, Qc);
  k_ltab<<<(257*NTAG + 127)/128, 128, 0, stream>>>(len_emb, fc_w, fc_b, Ltab);
  k_labels<<<1, S_, 0, stream>>>(target, runid, startf, endf, acc);

  // per-pair logits + loss
  k_pairs<<<S_, S_, 0, stream>>>(Zc, Qc, G, Ltab, runid, startf, endf, target, out, acc);
  k_finalize<<<1, 1, 0, stream>>>(acc, out);
}
